// TemporalCrossTransformer_55465207660692
// MI455X (gfx1250) — compile-verified, matrix operand order checked by reading, not run on hardware
//
#include <hip/hip_runtime.h>
#include <hip/hip_bf16.h>

// ---------------------------------------------------------------------------
// TemporalCrossTransformer forward for MI455X (gfx1250), wave32 + WMMA bf16
// + Tensor Data Mover staging (tensor_load_to_lds / s_wait_tensorcnt).
//
// Dimensions (fixed by the reference):
//   WAY=5 SHOT=5 (25 support), NQ=200 queries, S=10, D_IN=2048, D=128
//   tuples: C(10,2)=45 (t=2) + C(10,3)=120 (t=3) -> C=165 (pad 176)
//   support (s,m) flattened: 5*176=880 (pad 896 so K is a multiple of 32)
//
// All GEMMs use v_wmma_f32_16x16x32_bf16 (f32 accumulate).
// Fragment layout (ISA 7.12.2, wave32):
//   A (16x32 bf16): lane L (0..15) -> row M=L, K = {kk..kk+7, kk+16..kk+23}
//                   lane L (16..31)-> row M=L-16, K = {kk+8..kk+15, kk+24..kk+31}
//   B mirrors A with N in place of M (B stored N-major, K contiguous).
//   C/D (16x16 f32): acc[r], lane L -> M = r + 8*(L>>4), N = L&15.
// ---------------------------------------------------------------------------

typedef __attribute__((ext_vector_type(16))) __bf16 bf16x16;
typedef __attribute__((ext_vector_type(8)))  float  f32x8;
typedef __attribute__((ext_vector_type(4)))  unsigned u32x4;
typedef __attribute__((ext_vector_type(8)))  int      i32x8;
typedef __attribute__((ext_vector_type(4)))  int      i32x4;

#define NS 25
#define NQ 200
#define NT 225
#define SEQ 10
#define DIN 2048
#define DD 128
#define CVALID 165
#define CPAD 176
#define SCVALID 880   // 5 shots * 176
#define SCP 896       // padded to multiple of 32
#define MROWS 2250    // NT*SEQ
#define MPAD 2256     // padded to multiple of 16
#define NPROJ 1280    // 10 weight parts * 128

__device__ __forceinline__ __bf16 f2bf(float f) {
    unsigned u; __builtin_memcpy(&u, &f, 4);
    unsigned r = u + 0x7FFFu + ((u >> 16) & 1u);   // round-to-nearest-even
    unsigned short h = (unsigned short)(r >> 16);
    __bf16 b; __builtin_memcpy(&b, &h, 2);
    return b;
}

__device__ __forceinline__ bf16x16 ldfrag(const __bf16* rowp, int kb) {
    // rowp = start of the K-contiguous row for this lane; kb = lane-adjusted K base.
    struct { uint4 lo, hi; } u;
    u.lo = *(const uint4*)(rowp + kb);        // K = kb .. kb+7
    u.hi = *(const uint4*)(rowp + kb + 16);   // K = kb+16 .. kb+23
    bf16x16 v; __builtin_memcpy(&v, &u, 32);
    return v;
}

__device__ __forceinline__ f32x8 wmma_bf16(bf16x16 a, bf16x16 b, f32x8 c) {
    return __builtin_amdgcn_wmma_f32_16x16x32_bf16(false, a, false, b, (short)0, c,
                                                   false, false);
}

// TDM: DMA `nelem2B` 2-byte elements of contiguous global memory into LDS.
// 1-D tile descriptor: count=1, data_size=2B, tile_dim1=0, type=2 ("image").
__device__ __forceinline__ void tdm_load_1d(const void* gptr, unsigned lds_off,
                                            unsigned nelem2B) {
    unsigned long long ga = (unsigned long long)(size_t)gptr;
    u32x4 g0;
    g0[0] = 1u;                                            // count = 1 (valid D#)
    g0[1] = lds_off;                                       // lds_addr
    g0[2] = (unsigned)(ga & 0xFFFFFFFFull);                // global_addr[31:0]
    g0[3] = (unsigned)((ga >> 32) & 0x01FFFFFFull)         // global_addr[56:32]
            | (2u << 30);                                  // type = 2
    i32x8 g1;
    g1[0] = (int)(1u << 16);           // workgroup_mask=0, data_size=1 (2 bytes)
    g1[1] = (int)(nelem2B << 16);      // tensor_dim0[15:0]
    g1[2] = (int)((nelem2B >> 16) | (1u << 16)); // tensor_dim0[31:16], tensor_dim1=1
    g1[3] = (int)(nelem2B << 16);      // tile_dim0
    g1[4] = 0;                         // tile_dim1 = 0 -> 1-D tile
    g1[5] = (int)nelem2B;              // tensor_dim0_stride
    g1[6] = 0;
    g1[7] = 0;
    i32x4 z4 = {};
#if __clang_major__ >= 23
    i32x8 z8 = {};
    __builtin_amdgcn_tensor_load_to_lds(g0, g1, z4, z4, z8, 0);
#else
    __builtin_amdgcn_tensor_load_to_lds(g0, g1, z4, z4, 0);
#endif
}

// ---------------------------------------------------------------------------
// small utility kernels
// ---------------------------------------------------------------------------
__global__ void k_tuples(int* tup2, int* tup3) {
    if (blockIdx.x == 0 && threadIdx.x == 0) {
        int c = 0;
        for (int i = 0; i < SEQ; ++i)
            for (int j = i + 1; j < SEQ; ++j) { tup2[c*2] = i; tup2[c*2+1] = j; ++c; }
        c = 0;
        for (int i = 0; i < SEQ; ++i)
            for (int j = i + 1; j < SEQ; ++j)
                for (int k = j + 1; k < SEQ; ++k) {
                    tup3[c*3] = i; tup3[c*3+1] = j; tup3[c*3+2] = k; ++c;
                }
    }
}

__global__ void k_zero32(unsigned* p, int n) {
    int i = blockIdx.x * blockDim.x + threadIdx.x;
    if (i < n) p[i] = 0u;
}

// X (support|queries) + positional encoding -> bf16, rows padded to MPAD
__global__ void __launch_bounds__(256) k_prep_x(const float* sup, const float* qry,
                                                __bf16* Xb) {
    int row = blockIdx.x;               // 0..MPAD-1
    __bf16* orow = Xb + (size_t)row * DIN;
    if (row >= MROWS) {
        for (int d = threadIdx.x; d < DIN; d += 256) orow[d] = f2bf(0.f);
        return;
    }
    int sample = row / SEQ, pos = row % SEQ;
    const float* src = (sample < NS) ? (sup + ((size_t)sample * SEQ + pos) * DIN)
                                     : (qry + ((size_t)(sample - NS) * SEQ + pos) * DIN);
    const float ninv = -logf(10000.f) / (float)DIN;
    for (int d = threadIdx.x; d < DIN; d += 256) {
        float freq = __expf((float)(d & ~1) * ninv);
        float ang  = (float)pos * freq;
        float pe   = ((d & 1) ? __cosf(ang) : __sinf(ang)) * 0.1f;
        orow[d] = f2bf(src[d] + pe);
    }
}

// Weight slices -> N-major bf16: WT[(w*128+n)*2048 + k] = W[part*2048+k][n]
// part order w=0..9: kw0p0,kw0p1, kw1p0,kw1p1,kw1p2, vw0p0,vw0p1, vw1p0,vw1p1,vw1p2
__global__ void __launch_bounds__(256) k_prep_w(const float* kw0, const float* vw0,
                                                const float* kw1, const float* vw1,
                                                __bf16* WT) {
    int gid = blockIdx.x * 256 + threadIdx.x;
    if (gid >= 10 * DD * DIN) return;
    int w   = gid / (DD * DIN);
    int rem = gid % (DD * DIN);
    int n = rem / DIN, k = rem % DIN;
    const float* src; int part;
    switch (w) {
        case 0: src = kw0; part = 0; break;
        case 1: src = kw0; part = 1; break;
        case 2: src = kw1; part = 0; break;
        case 3: src = kw1; part = 1; break;
        case 4: src = kw1; part = 2; break;
        case 5: src = vw0; part = 0; break;
        case 6: src = vw0; part = 1; break;
        case 7: src = vw1; part = 0; break;
        case 8: src = vw1; part = 1; break;
        default: src = vw1; part = 2; break;
    }
    WT[(size_t)gid] = f2bf(src[((size_t)(part * DIN + k)) * DD + n]);
}

// ---------------------------------------------------------------------------
// projection GEMM: proj[2256][1280] = Xb[2256][2048] @ WT^T   (bf16 WMMA)
// one wave computes a 16x64 output tile; fragments are double-buffered so
// VMEM for step k+1 overlaps the WMMAs of step k.
// ---------------------------------------------------------------------------
__global__ void __launch_bounds__(256) k_proj_gemm(const __bf16* __restrict__ Xb,
                                                   const __bf16* __restrict__ WT,
                                                   float* __restrict__ proj) {
    const int NGROUPS = NPROJ / 64;                 // 20
    const int MT = MPAD / 16;                       // 141
    int wid = blockIdx.x * 8 + (threadIdx.x >> 5);
    if (wid >= MT * NGROUPS) return;
    int mt = wid / NGROUPS, ng = wid % NGROUPS;
    int lane = threadIdx.x & 31;
    int l15 = lane & 15;
    int kh  = (lane >> 4) * 8;
    int mb  = (lane >> 4) * 8;
    int n0  = ng * 64;

    const __bf16* arow = Xb + (size_t)(mt * 16 + l15) * DIN;
    const __bf16* b0 = WT + (size_t)(n0 +  0 + l15) * DIN;
    const __bf16* b1 = WT + (size_t)(n0 + 16 + l15) * DIN;
    const __bf16* b2 = WT + (size_t)(n0 + 32 + l15) * DIN;
    const __bf16* b3 = WT + (size_t)(n0 + 48 + l15) * DIN;

    f32x8 acc0 = {}, acc1 = {}, acc2 = {}, acc3 = {};
    bf16x16 a  = ldfrag(arow, kh);
    bf16x16 f0 = ldfrag(b0, kh);
    bf16x16 f1 = ldfrag(b1, kh);
    bf16x16 f2 = ldfrag(b2, kh);
    bf16x16 f3 = ldfrag(b3, kh);
    for (int kk = 32; kk < DIN; kk += 32) {
        int kb = kk + kh;
        __builtin_prefetch(arow + kb + 256, 0, 1);     // global_prefetch_b8
        bf16x16 an = ldfrag(arow, kb);
        bf16x16 g0 = ldfrag(b0, kb);
        bf16x16 g1 = ldfrag(b1, kb);
        bf16x16 g2 = ldfrag(b2, kb);
        bf16x16 g3 = ldfrag(b3, kb);
        acc0 = wmma_bf16(a, f0, acc0);
        acc1 = wmma_bf16(a, f1, acc1);
        acc2 = wmma_bf16(a, f2, acc2);
        acc3 = wmma_bf16(a, f3, acc3);
        a = an; f0 = g0; f1 = g1; f2 = g2; f3 = g3;
    }
    acc0 = wmma_bf16(a, f0, acc0);
    acc1 = wmma_bf16(a, f1, acc1);
    acc2 = wmma_bf16(a, f2, acc2);
    acc3 = wmma_bf16(a, f3, acc3);

    int m0 = mt * 16 + mb;
    for (int r = 0; r < 8; ++r) {
        float* orow = proj + (size_t)(m0 + r) * NPROJ + n0;
        orow[ 0 + l15] = acc0[r];
        orow[16 + l15] = acc1[r];
        orow[32 + l15] = acc2[r];
        orow[48 + l15] = acc3[r];
    }
}

// ---------------------------------------------------------------------------
// tuple assembly + bias + LayerNorm -> Qk/Qv (queries), Sk/Svt (support, class
// ordered).  One block per (sample, cpad), 128 threads = one per d.
// ---------------------------------------------------------------------------
__global__ void __launch_bounds__(128) k_assemble_ln(
        const float* __restrict__ proj, const int* tup2, const int* tup3,
        const float* kb0, const float* vb0, const float* kb1, const float* vb1,
        const float* gamma, const float* beta, const int* labels,
        __bf16* Qk, __bf16* Qvb, float* Qvf, __bf16* Sk, __bf16* Svt) {
    int bx = blockIdx.x;
    int sample = bx / CPAD, c = bx % CPAD;
    int d = threadIdx.x;
    __shared__ float red[128];
    bool valid = (c < CVALID);

    float kval = 0.f, vval = 0.f;
    if (valid) {
        if (c < 45) {
            int i = tup2[c*2], j = tup2[c*2+1];
            const float* r0 = proj + (size_t)(sample*SEQ + i) * NPROJ;
            const float* r1 = proj + (size_t)(sample*SEQ + j) * NPROJ;
            kval = r0[0*DD + d] + r1[1*DD + d] + kb0[d];
            vval = r0[5*DD + d] + r1[6*DD + d] + vb0[d];
        } else {
            int cc = c - 45;
            int i = tup3[cc*3], j = tup3[cc*3+1], k = tup3[cc*3+2];
            const float* r0 = proj + (size_t)(sample*SEQ + i) * NPROJ;
            const float* r1 = proj + (size_t)(sample*SEQ + j) * NPROJ;
            const float* r2 = proj + (size_t)(sample*SEQ + k) * NPROJ;
            kval = r0[2*DD + d] + r1[3*DD + d] + r2[4*DD + d] + kb1[d];
            vval = r0[7*DD + d] + r1[8*DD + d] + r2[9*DD + d] + vb1[d];
        }
    }
    auto block_sum = [&](float x) -> float {
        red[d] = x; __syncthreads();
        for (int s = 64; s > 0; s >>= 1) {
            if (d < s) red[d] += red[d + s];
            __syncthreads();
        }
        float r = red[0]; __syncthreads();
        return r;
    };
    float kmu  = block_sum(kval) * (1.f / DD);
    float kvar = block_sum((kval - kmu) * (kval - kmu)) * (1.f / DD);
    float kn   = (kval - kmu) * rsqrtf(kvar + 1e-5f) * gamma[d] + beta[d];
    float vmu  = block_sum(vval) * (1.f / DD);
    float vvar = block_sum((vval - vmu) * (vval - vmu)) * (1.f / DD);
    float vn   = (vval - vmu) * rsqrtf(vvar + 1e-5f) * gamma[d] + beta[d];
    if (!valid) { kn = 0.f; vn = 0.f; }

    if (sample >= NS) {
        int q = sample - NS;
        Qk [(size_t)(q*CPAD + c) * DD + d] = f2bf(kn);
        Qvb[(size_t)(q*CPAD + c) * DD + d] = f2bf(vn);
        Qvf[(size_t)(q*CPAD + c) * DD + d] = vn;
    } else {
        int cls = labels[sample];
        int slot = 0;
        for (int p = 0; p < sample; ++p) if (labels[p] == cls) ++slot;
        int sc = slot * CPAD + c;
        Sk [(size_t)(cls*SCP + sc) * DD + d]  = f2bf(kn);
        Svt[(size_t)(cls*DD + d) * SCP + sc]  = f2bf(vn);
    }
}

__global__ void __launch_bounds__(256) k_qnorm(const float* Qvf, float* qn) {
    int q = blockIdx.x;
    for (int m = threadIdx.x; m < CPAD; m += 256) {
        float s = 0.f;
        if (m < CVALID) {
            const float* r = Qvf + (size_t)(q*CPAD + m) * DD;
            for (int d = 0; d < DD; ++d) s += r[d] * r[d];
        }
        qn[q*CPAD + m] = s;
    }
}

// ---------------------------------------------------------------------------
// fused attention: one block per (q, cls).  Loops over 11 n-tiles:
//   TDM stages the 16x128 Qk tile into LDS (one 4KB contiguous block),
//   scores tile (16 x 896) = Qk_tile @ Sk^T  (WMMA, A-frags hoisted from LDS)
//   masked softmax over the flattened (shot,m) axis
//   proto tile (16 x 128)  = attn @ CV       (WMMA) -> protoF/protoB
// ---------------------------------------------------------------------------
__global__ void __launch_bounds__(256) k_attention(
        const __bf16* __restrict__ Qk, const __bf16* __restrict__ Sk,
        const __bf16* __restrict__ Svt,
        float* __restrict__ protoF, __bf16* __restrict__ protoB) {
    int q = blockIdx.x / 5, cls = blockIdx.x % 5;
    int lane = threadIdx.x & 31, wave = threadIdx.x >> 5;
    int l15 = lane & 15;
    int kh = (lane >> 4) * 8, mb = (lane >> 4) * 8;
    __shared__ float St[16 * SCP];                     // 57344 B
    __shared__ __align__(16) __bf16 Alds[16 * DD];     // 4096 B (Qk tile)
    const float scale = 0.088388347648318447f;         // 1/sqrt(128)
    const __bf16* qkb = Qk  + (size_t)q   * CPAD * DD;
    const __bf16* skb = Sk  + (size_t)cls * SCP  * DD;
    const __bf16* svb = Svt + (size_t)cls * DD   * SCP;
    float* pf  = protoF + (size_t)blockIdx.x * CPAD * DD;
    __bf16* pb = protoB + (size_t)blockIdx.x * CPAD * DD;

    for (int nt = 0; nt < CPAD / 16; ++nt) {
        int n0 = nt * 16;
        // ---- stage A tile (16 rows x 128 d, contiguous 4KB) via TDM
        if (wave == 0) {
            tdm_load_1d(qkb + (size_t)n0 * DD,
                        (unsigned)(size_t)(&Alds[0]), 16 * DD);
            __builtin_amdgcn_s_wait_tensorcnt(0);
        }
        __syncthreads();
        // ---- scores GEMM: 56 N-tiles over (shot,m), K = 128
        {
            const __bf16* alds = &Alds[l15 * DD];
            bf16x16 a0 = ldfrag(alds,  0 + kh);
            bf16x16 a1 = ldfrag(alds, 32 + kh);
            bf16x16 a2 = ldfrag(alds, 64 + kh);
            bf16x16 a3 = ldfrag(alds, 96 + kh);
            for (int ct = wave; ct < SCP / 16; ct += 8) {
                const __bf16* brow = skb + (size_t)(ct * 16 + l15) * DD;
                f32x8 acc = {};
                acc = wmma_bf16(a0, ldfrag(brow,  0 + kh), acc);
                acc = wmma_bf16(a1, ldfrag(brow, 32 + kh), acc);
                acc = wmma_bf16(a2, ldfrag(brow, 64 + kh), acc);
                acc = wmma_bf16(a3, ldfrag(brow, 96 + kh), acc);
                for (int r = 0; r < 8; ++r)
                    St[(r + mb) * SCP + ct * 16 + l15] = acc[r] * scale;
            }
        }
        __syncthreads();
        // ---- masked softmax: 16 rows, 16 threads per row
        {
            int rrow = threadIdx.x >> 4, j = threadIdx.x & 15;
            float mx = -1e30f;
            for (int col = j; col < SCVALID; col += 16)
                if ((col % CPAD) < CVALID) mx = fmaxf(mx, St[rrow * SCP + col]);
            for (int s = 8; s >= 1; s >>= 1) mx = fmaxf(mx, __shfl_xor(mx, s, 16));
            float sum = 0.f;
            for (int col = j; col < SCVALID; col += 16) {
                float e = 0.f;
                if ((col % CPAD) < CVALID) e = __expf(St[rrow * SCP + col] - mx);
                St[rrow * SCP + col] = e;
                sum += e;
            }
            for (int s = 8; s >= 1; s >>= 1) sum += __shfl_xor(sum, s, 16);
            float inv = 1.f / sum;
            for (int col = j; col < SCP; col += 16)
                St[rrow * SCP + col] = (col < SCVALID) ? St[rrow * SCP + col] * inv : 0.f;
        }
        __syncthreads();
        // ---- proto GEMM: each wave owns one 16-wide d-tile, K = 896
        {
            int d0 = wave * 16;
            const __bf16* brow = svb + (size_t)(d0 + l15) * SCP;
            const float*  afr  = St + l15 * SCP;
            f32x8 acc = {};
            for (int kk = 0; kk < SCP; kk += 32) {
                int kb = kk + kh;
                bf16x16 a;
                for (int e = 0; e < 8; ++e) a[e]     = f2bf(afr[kb + e]);
                for (int e = 0; e < 8; ++e) a[8 + e] = f2bf(afr[kb + 16 + e]);
                acc = wmma_bf16(a, ldfrag(brow, kb), acc);
            }
            for (int r = 0; r < 8; ++r) {
                int m = n0 + r + mb;
                pf[(size_t)m * DD + d0 + l15] = acc[r];
                pb[(size_t)m * DD + d0 + l15] = f2bf(acc[r]);
            }
        }
        __syncthreads();
    }
}

// proto row norms + L2 column:  -(||Qv - proto||^2) / C
__global__ void __launch_bounds__(256) k_pnorm_l2(const float* __restrict__ protoF,
                                                  const float* __restrict__ Qvf,
                                                  float* pn, float* l2out) {
    int q = blockIdx.x / 5;
    const float* P  = protoF + (size_t)blockIdx.x * CPAD * DD;
    const float* Qv = Qvf    + (size_t)q          * CPAD * DD;
    __shared__ float red[256];
    for (int m = threadIdx.x; m < CPAD; m += 256) {
        float s = 0.f;
        if (m < CVALID) {
            const float* r = P + (size_t)m * DD;
            for (int d = 0; d < DD; ++d) s += r[d] * r[d];
        }
        pn[blockIdx.x * CPAD + m] = s;
    }
    float part = 0.f;
    for (int e = threadIdx.x; e < CVALID * DD; e += 256) {
        float df = Qv[e] - P[e];       // first 165*128 elements are the valid block
        part += df * df;
    }
    red[threadIdx.x] = part; __syncthreads();
    for (int s = 128; s > 0; s >>= 1) {
        if (threadIdx.x < s) red[threadIdx.x] += red[threadIdx.x + s];
        __syncthreads();
    }
    if (threadIdx.x == 0) l2out[blockIdx.x] = -red[0] / (float)CVALID;
}

// cdist: M[n][m] = sqrt(max(||qv_n||^2 + ||p_m||^2 - 2 qv_n.p_m, 1e-12))  (WMMA)
// a wave owns one mi row-tile and reuses its 4 A-fragments across all ni tiles.
__global__ void __launch_bounds__(256) k_cdist(const __bf16* __restrict__ Qvb,
                                               const __bf16* __restrict__ protoB,
                                               const float* qn, const float* pn,
                                               float* __restrict__ Md) {
    int q = blockIdx.x / 5;
    int lane = threadIdx.x & 31, wave = threadIdx.x >> 5;
    int l15 = lane & 15;
    int kh = (lane >> 4) * 8, mb = (lane >> 4) * 8;
    const __bf16* A = Qvb    + (size_t)q          * CPAD * DD;
    const __bf16* B = protoB + (size_t)blockIdx.x * CPAD * DD;
    float* M = Md + (size_t)blockIdx.x * CPAD * CPAD;
    const float* qnr = qn + q * CPAD;
    const float* pnr = pn + blockIdx.x * CPAD;
    for (int mi = wave; mi < CPAD / 16; mi += 8) {
        const __bf16* arow = A + (size_t)(mi * 16 + l15) * DD;
        bf16x16 a0 = ldfrag(arow,  0 + kh);
        bf16x16 a1 = ldfrag(arow, 32 + kh);
        bf16x16 a2 = ldfrag(arow, 64 + kh);
        bf16x16 a3 = ldfrag(arow, 96 + kh);
        for (int ni = 0; ni < CPAD / 16; ++ni) {
            const __bf16* brow = B + (size_t)(ni * 16 + l15) * DD;
            f32x8 acc = {};
            acc = wmma_bf16(a0, ldfrag(brow,  0 + kh), acc);
            acc = wmma_bf16(a1, ldfrag(brow, 32 + kh), acc);
            acc = wmma_bf16(a2, ldfrag(brow, 64 + kh), acc);
            acc = wmma_bf16(a3, ldfrag(brow, 96 + kh), acc);
            for (int r = 0; r < 8; ++r) {
                int n = mi * 16 + r + mb, m = ni * 16 + l15;
                float d2 = qnr[n] + pnr[m] - 2.f * acc[r];
                M[(size_t)n * CPAD + m] =
                    (n < CVALID && m < CVALID) ? sqrtf(fmaxf(d2, 1e-12f)) : 0.f;
            }
        }
    }
}

// Sinkhorn (20 iterations) + <P, M>; one block per (q, cls)
__global__ void __launch_bounds__(256) k_sinkhorn(const float* __restrict__ Md,
                                                  float* __restrict__ Pb,
                                                  float* otout) {
    const float* M = Md + (size_t)blockIdx.x * CPAD * CPAD;
    float* P       = Pb + (size_t)blockIdx.x * CPAD * CPAD;
    __shared__ float red[256];
    __shared__ float rs[CVALID];
    int tid = threadIdx.x;
    float part = 0.f;
    for (int e = tid; e < CVALID * CVALID; e += 256) {
        int n = e / CVALID, m = e % CVALID;
        float v = __expf(-M[n * CPAD + m]);
        P[n * CPAD + m] = v; part += v;
    }
    red[tid] = part; __syncthreads();
    for (int s = 128; s > 0; s >>= 1) {
        if (tid < s) red[tid] += red[tid + s];
        __syncthreads();
    }
    float inv = 1.f / red[0];
    __syncthreads();
    for (int e = tid; e < CVALID * CVALID; e += 256) {
        int n = e / CVALID, m = e % CVALID;
        P[n * CPAD + m] *= inv;
    }
    __syncthreads();
    const float rmarg = 1.f / (float)CVALID;
    for (int it = 0; it < 20; ++it) {
        for (int n = tid; n < CVALID; n += 256) {
            float s = 0.f;
            const float* pr = P + n * CPAD;
            for (int m = 0; m < CVALID; ++m) s += pr[m];
            rs[n] = rmarg / s;
        }
        __syncthreads();
        for (int e = tid; e < CVALID * CVALID; e += 256) {
            int n = e / CVALID, m = e % CVALID;
            P[n * CPAD + m] *= rs[n];
        }
        __syncthreads();
        for (int m = tid; m < CVALID; m += 256) {
            float s = 0.f;
            for (int n = 0; n < CVALID; ++n) s += P[n * CPAD + m];
            rs[m] = rmarg / s;
        }
        __syncthreads();
        for (int e = tid; e < CVALID * CVALID; e += 256) {
            int n = e / CVALID, m = e % CVALID;
            P[n * CPAD + m] *= rs[m];
        }
        __syncthreads();
    }
    part = 0.f;
    for (int e = tid; e < CVALID * CVALID; e += 256) {
        int n = e / CVALID, m = e % CVALID;
        part += P[n * CPAD + m] * M[n * CPAD + m];
    }
    red[tid] = part; __syncthreads();
    for (int s = 128; s > 0; s >>= 1) {
        if (tid < s) red[tid] += red[tid + s];
        __syncthreads();
    }
    if (tid == 0) otout[blockIdx.x] = -red[0];
}

// fusion head + batch-norm over queries; writes the (200 x 15) output
__global__ void __launch_bounds__(256) k_fusion(const float* l2, const float* ot,
                                                const float* fw, const float* fb,
                                                const float* bg, const float* bb,
                                                float* out) {
    __shared__ float fus[NQ * 5];
    __shared__ float mu[5], iv[5];
    int tid = threadIdx.x;
    for (int e = tid; e < NQ * 5; e += 256) {
        int q = e / 5, w = e % 5;
        float s = fb[w];
        for (int j = 0; j < 5; ++j) s += l2[q*5 + j] * fw[j*5 + w];
        for (int j = 0; j < 5; ++j) s += ot[q*5 + j] * fw[(5 + j)*5 + w];
        fus[e] = s;
    }
    __syncthreads();
    if (tid < 5) {
        float s = 0.f, s2 = 0.f;
        for (int q = 0; q < NQ; ++q) { float v = fus[q*5 + tid]; s += v; s2 += v * v; }
        float m = s / (float)NQ;
        mu[tid] = m;
        iv[tid] = rsqrtf(s2 / (float)NQ - m * m + 1e-5f);
    }
    __syncthreads();
    for (int e = tid; e < NQ * 5; e += 256) {
        int q = e / 5, w = e % 5;
        out[q*15 + w]      = l2[e];
        out[q*15 + 5 + w]  = ot[e];
        out[q*15 + 10 + w] = (fus[e] - mu[w]) * iv[w] * bg[w] + bb[w];
    }
}

// ---------------------------------------------------------------------------
extern "C" void kernel_launch(void* const* d_in, const int* in_sizes, int n_in,
                              void* d_out, int out_size, void* d_ws, size_t ws_size,
                              hipStream_t stream) {
    (void)in_sizes; (void)n_in; (void)out_size; (void)ws_size;
    const float* support  = (const float*)d_in[0];
    const float* queries  = (const float*)d_in[1];
    const float* kw0      = (const float*)d_in[2];
    const float* kb0      = (const float*)d_in[3];
    const float* vw0      = (const float*)d_in[4];
    const float* vb0      = (const float*)d_in[5];
    const float* kw1      = (const float*)d_in[6];
    const float* kb1      = (const float*)d_in[7];
    const float* vw1      = (const float*)d_in[8];
    const float* vb1      = (const float*)d_in[9];
    const float* gamma_k  = (const float*)d_in[10];
    const float* beta_k   = (const float*)d_in[11];
    const float* fusion_w = (const float*)d_in[12];
    const float* fusion_b = (const float*)d_in[13];
    const float* bn_gamma = (const float*)d_in[14];
    const float* bn_beta  = (const float*)d_in[15];
    const int*   labels   = (const int*)d_in[16];

    char* base = (char*)d_ws;
    size_t off = 0;
    auto take = [&](size_t bytes) -> size_t {
        size_t r = off;
        off = (off + bytes + 255) & ~(size_t)255;
        return r;
    };
    size_t oXB   = take((size_t)MPAD * DIN * 2);
    size_t oWT   = take((size_t)10 * DD * DIN * 2);
    size_t oPROJ = take((size_t)MPAD * NPROJ * 4);
    size_t oT2   = take(45 * 2 * 4);
    size_t oT3   = take(120 * 3 * 4);
    size_t oSK   = take((size_t)5 * SCP * DD * 2);
    size_t oSVT  = take((size_t)5 * DD * SCP * 2);
    size_t oQK   = take((size_t)NQ * CPAD * DD * 2);
    size_t oQVB  = take((size_t)NQ * CPAD * DD * 2);
    size_t oQVF  = take((size_t)NQ * CPAD * DD * 4);
    size_t oQN   = take((size_t)NQ * CPAD * 4);
    size_t oPF   = take((size_t)NQ * 5 * CPAD * DD * 4);
    size_t oPB   = take((size_t)NQ * 5 * CPAD * DD * 2);
    size_t oPN   = take((size_t)NQ * 5 * CPAD * 4);
    size_t oL2   = take((size_t)NQ * 5 * 4);
    size_t oOT   = take((size_t)NQ * 5 * 4);
    size_t oMD   = take((size_t)NQ * 5 * CPAD * CPAD * 4);
    size_t oPP   = take((size_t)NQ * 5 * CPAD * CPAD * 4);

    __bf16* Xb  = (__bf16*)(base + oXB);
    __bf16* WT  = (__bf16*)(base + oWT);
    float*  prj = (float*)(base + oPROJ);
    int* tup2   = (int*)(base + oT2);
    int* tup3   = (int*)(base + oT3);
    __bf16* Sk  = (__bf16*)(base + oSK);
    __bf16* Svt = (__bf16*)(base + oSVT);
    __bf16* Qk  = (__bf16*)(base + oQK);
    __bf16* Qvb = (__bf16*)(base + oQVB);
    float*  Qvf = (float*)(base + oQVF);
    float*  qn  = (float*)(base + oQN);
    float*  pF  = (float*)(base + oPF);
    __bf16* pB  = (__bf16*)(base + oPB);
    float*  pn  = (float*)(base + oPN);
    float*  l2v = (float*)(base + oL2);
    float*  otv = (float*)(base + oOT);
    float*  Md  = (float*)(base + oMD);
    float*  Pp  = (float*)(base + oPP);

    k_tuples<<<1, 32, 0, stream>>>(tup2, tup3);
    {   // zero Sk / Svt (pad rows 880..895 are never written by assembly)
        int n = (int)((size_t)5 * SCP * DD * 2 / 4);
        k_zero32<<<(n + 255) / 256, 256, 0, stream>>>((unsigned*)Sk, n);
        k_zero32<<<(n + 255) / 256, 256, 0, stream>>>((unsigned*)Svt, n);
    }
    k_prep_x<<<MPAD, 256, 0, stream>>>(support, queries, Xb);
    k_prep_w<<<(10 * DD * DIN + 255) / 256, 256, 0, stream>>>(kw0, vw0, kw1, vw1, WT);

    {   // projection GEMM
        int jobs = (MPAD / 16) * (NPROJ / 64);
        k_proj_gemm<<<(jobs + 7) / 8, 256, 0, stream>>>(Xb, WT, prj);
    }
    k_assemble_ln<<<NT * CPAD, 128, 0, stream>>>(prj, tup2, tup3, kb0, vb0, kb1, vb1,
                                                 gamma_k, beta_k, labels,
                                                 Qk, Qvb, Qvf, Sk, Svt);
    k_qnorm<<<NQ, 256, 0, stream>>>(Qvf, qn);
    k_attention<<<NQ * 5, 256, 0, stream>>>(Qk, Sk, Svt, pF, pB);
    k_pnorm_l2<<<NQ * 5, 256, 0, stream>>>(pF, Qvf, pn, l2v);
    k_cdist<<<NQ * 5, 256, 0, stream>>>(Qvb, pB, qn, pn, Md);
    k_sinkhorn<<<NQ * 5, 256, 0, stream>>>(Md, Pp, otv);
    k_fusion<<<1, 256, 0, stream>>>(l2v, otv, fusion_w, fusion_b, bn_gamma, bn_beta,
                                    (float*)d_out);
}
